// MultiViewDepthConsistency_57123065037497
// MI455X (gfx1250) — matrix operand order repaired and checked by reading, not compiled, not run on hardware
//
#include <hip/hip_runtime.h>
#include <hip/hip_bf16.h>
#include <math.h>

// ---------------------------------------------------------------------------
// MultiViewDepthConsistency for MI455X (gfx1250, wave32)
//  - confidence conv: v_wmma_f32_16x16x32_f16 (im2col GEMM, K=9 padded to 32)
//  - weights staged to LDS via GLOBAL_LOAD_ASYNC_TO_LDS (gfx1250 async path)
//  - geometric loss: f32 VALU + bilinear gathers + wave shfl reduction + atomics
// ---------------------------------------------------------------------------

#define HH 384
#define WW 384
#define BB 2
#define NN 4
#define NPAIRS 12
#define DELTA_H 0.1f

#ifndef __has_builtin
#define __has_builtin(x) 0
#endif

#if __has_builtin(__builtin_amdgcn_global_load_async_to_lds_b32) && \
    __has_builtin(__builtin_amdgcn_s_wait_asynccnt)
#define USE_ASYNC_LDS 1
#else
#define USE_ASYNC_LDS 0
#endif

typedef __attribute__((ext_vector_type(16))) _Float16 v16h;
typedef __attribute__((ext_vector_type(8)))  float    v8f;

typedef __attribute__((address_space(1))) int gint_t;   // global b32
typedef __attribute__((address_space(3))) int lint_t;   // LDS b32

__constant__ int c_ps[NPAIRS] = {0,1,0,2,0,3,1,2,1,3,2,3};
__constant__ int c_pt[NPAIRS] = {1,0,2,0,3,0,2,1,3,1,3,2};

// LDS weight cache layout: w1[288] | b1[32] | w2[32] | b2[1]
#define LW_W1 0
#define LW_B1 288
#define LW_W2 320
#define LW_B2 352
#define LW_TOT 353

// ---------------------------------------------------------------------------
// Kernel 1: confidence = sigmoid(conv1x1(relu(conv3x3(depth))))
// One wave per 16-pixel row segment; two WMMAs cover the 32 output channels.
// ---------------------------------------------------------------------------
__global__ __launch_bounds__(256) void conf_conv_wmma(
    const float* __restrict__ depth,   // (B*N, H, W)
    const float* __restrict__ w1,      // (32, 1, 3, 3)
    const float* __restrict__ b1,      // (32)
    const float* __restrict__ w2,      // (1, 32, 1, 1)
    const float* __restrict__ b2,      // (1)
    float* __restrict__ conf)          // (B*N, H, W)
{
    __shared__ float swb[LW_TOT];

    const int tid    = threadIdx.x;
    const int lane   = tid & 31;
    const int waveId = tid >> 5;
    const long tile  = (long)blockIdx.x * 8 + waveId;   // 16 pixels per tile
    const long pix0  = tile * 16;

    const int img = (int)(pix0 / (HH * WW));
    const int rem = (int)(pix0 % (HH * WW));
    const int y   = rem / WW;
    const int x0  = rem % WW;                           // multiple of 16
    const float* dimg = depth + (size_t)img * HH * WW;

    // Kick off prefetch of the center depth row while weights are staged.
    __builtin_prefetch(dimg + (size_t)y * WW + x0, 0, 3);

    // ---- stage weights into LDS (async on gfx1250-native toolchains) ----
    if (tid < LW_TOT) {
        const float* src = (tid < LW_B1) ? (w1 + tid)
                         : (tid < LW_W2) ? (b1 + (tid - LW_B1))
                         : (tid < LW_B2) ? (w2 + (tid - LW_W2))
                                         : b2;
#if USE_ASYNC_LDS
        __builtin_amdgcn_global_load_async_to_lds_b32(
            (gint_t*)src, (lint_t*)&swb[tid], 0, 0);
#else
        swb[tid] = *src;
#endif
    }
#if USE_ASYNC_LDS
    __builtin_amdgcn_s_wait_asynccnt(0);
#endif
    __syncthreads();

    const int m = lane & 15;      // pixel within tile (A row) == channel (B col)
    const int x = x0 + m;

    // ---- A matrix: 16x32 f16, K = 9-tap patch zero-padded to 32 ----
    // lanes 0-15: halves 0..7 = K 0..7 ; lanes 16-31: half 0 = K 8
    v16h a = {};
    if (lane < 16) {
#pragma unroll
        for (int k = 0; k < 8; ++k) {
            const int yy = y + k / 3 - 1;
            const int xx = x + k % 3 - 1;
            float v = 0.0f;
            if (yy >= 0 && yy < HH && xx >= 0 && xx < WW)
                v = dimg[yy * WW + xx];
            a[k] = (_Float16)v;
        }
    } else {
        const int yy = y + 1, xx = x + 1;
        float v = 0.0f;
        if (yy >= 0 && yy < HH && xx >= 0 && xx < WW)
            v = dimg[yy * WW + xx];
        a[0] = (_Float16)v;
    }

    // ---- B matrices: 32x16 f16 (K x N). lanes 0-15: halves 0..7 = K 0..7;
    // lanes 16-31: half 0 = K 8 (rest zero, same layout as A). ----
    v16h bmat0 = {};
    v16h bmat1 = {};
    if (lane < 16) {
#pragma unroll
        for (int k = 0; k < 8; ++k) {
            bmat0[k] = (_Float16)swb[LW_W1 + m * 9 + k];          // channels 0..15
            bmat1[k] = (_Float16)swb[LW_W1 + (16 + m) * 9 + k];   // channels 16..31
        }
    } else {
        bmat0[0] = (_Float16)swb[LW_W1 + m * 9 + 8];
        bmat1[0] = (_Float16)swb[LW_W1 + (16 + m) * 9 + 8];
    }

    v8f c0 = {};
    v8f c1 = {};
    c0 = __builtin_amdgcn_wmma_f32_16x16x32_f16(false, a, false, bmat0,
                                                (short)0, c0, false, false);
    c1 = __builtin_amdgcn_wmma_f32_16x16x32_f16(false, a, false, bmat1,
                                                (short)0, c1, false, false);

    // ---- bias + ReLU + 1x1 conv (channel reduction) + sigmoid ----
    const float b1lo = swb[LW_B1 + m],  b1hi = swb[LW_B1 + 16 + m];
    const float w2lo = swb[LW_W2 + m],  w2hi = swb[LW_W2 + 16 + m];
    const float bias2 = swb[LW_B2];

#pragma unroll
    for (int r = 0; r < 8; ++r) {
        float flo = c0[r] + b1lo; flo = flo > 0.0f ? flo : 0.0f;
        float fhi = c1[r] + b1hi; fhi = fhi > 0.0f ? fhi : 0.0f;
        float t = flo * w2lo + fhi * w2hi;
        // reduce over 16-lane half (channels); halves stay independent
        t += __shfl_xor(t, 1, 32);
        t += __shfl_xor(t, 2, 32);
        t += __shfl_xor(t, 4, 32);
        t += __shfl_xor(t, 8, 32);
        if ((lane & 15) == 0) {
            const int mpix = (lane < 16) ? r : (8 + r);  // lane0 -> M=r, lane16 -> M=8+r
            const float g = t + bias2;
            conf[pix0 + mpix] = 1.0f / (1.0f + __expf(-g));
        }
    }
}

// ---------------------------------------------------------------------------
// 4x4 inverse (adjugate), row-major
// ---------------------------------------------------------------------------
__device__ void inv4x4(const float* m, float* o)
{
    float inv[16];
    inv[0]  =  m[5]*m[10]*m[15] - m[5]*m[11]*m[14] - m[9]*m[6]*m[15] + m[9]*m[7]*m[14] + m[13]*m[6]*m[11] - m[13]*m[7]*m[10];
    inv[4]  = -m[4]*m[10]*m[15] + m[4]*m[11]*m[14] + m[8]*m[6]*m[15] - m[8]*m[7]*m[14] - m[12]*m[6]*m[11] + m[12]*m[7]*m[10];
    inv[8]  =  m[4]*m[9]*m[15]  - m[4]*m[11]*m[13] - m[8]*m[5]*m[15] + m[8]*m[7]*m[13] + m[12]*m[5]*m[11] - m[12]*m[7]*m[9];
    inv[12] = -m[4]*m[9]*m[14]  + m[4]*m[10]*m[13] + m[8]*m[5]*m[14] - m[8]*m[6]*m[13] - m[12]*m[5]*m[10] + m[12]*m[6]*m[9];
    inv[1]  = -m[1]*m[10]*m[15] + m[1]*m[11]*m[14] + m[9]*m[2]*m[15] - m[9]*m[3]*m[14] - m[13]*m[2]*m[11] + m[13]*m[3]*m[10];
    inv[5]  =  m[0]*m[10]*m[15] - m[0]*m[11]*m[14] - m[8]*m[2]*m[15] + m[8]*m[3]*m[14] + m[12]*m[2]*m[11] - m[12]*m[3]*m[10];
    inv[9]  = -m[0]*m[9]*m[15]  + m[0]*m[11]*m[13] + m[8]*m[1]*m[15] - m[8]*m[3]*m[13] - m[12]*m[1]*m[11] + m[12]*m[3]*m[9];
    inv[13] =  m[0]*m[9]*m[14]  - m[0]*m[10]*m[13] - m[8]*m[1]*m[14] + m[8]*m[2]*m[13] + m[12]*m[1]*m[10] - m[12]*m[2]*m[9];
    inv[2]  =  m[1]*m[6]*m[15]  - m[1]*m[7]*m[14]  - m[5]*m[2]*m[15] + m[5]*m[3]*m[14] + m[13]*m[2]*m[7]  - m[13]*m[3]*m[6];
    inv[6]  = -m[0]*m[6]*m[15]  + m[0]*m[7]*m[14]  + m[4]*m[2]*m[15] - m[4]*m[3]*m[14] - m[12]*m[2]*m[7]  + m[12]*m[3]*m[6];
    inv[10] =  m[0]*m[5]*m[15]  - m[0]*m[7]*m[13]  - m[4]*m[1]*m[15] + m[4]*m[3]*m[13] + m[12]*m[1]*m[7]  - m[12]*m[3]*m[5];
    inv[14] = -m[0]*m[5]*m[14]  + m[0]*m[6]*m[13]  + m[4]*m[1]*m[14] - m[4]*m[2]*m[13] - m[12]*m[1]*m[6]  + m[12]*m[2]*m[5];
    inv[3]  = -m[1]*m[6]*m[11]  + m[1]*m[7]*m[10]  + m[5]*m[2]*m[11] - m[5]*m[3]*m[10] - m[9]*m[2]*m[7]   + m[9]*m[3]*m[6];
    inv[7]  =  m[0]*m[6]*m[11]  - m[0]*m[7]*m[10]  - m[4]*m[2]*m[11] + m[4]*m[3]*m[10] + m[8]*m[2]*m[7]   - m[8]*m[3]*m[6];
    inv[11] = -m[0]*m[5]*m[11]  + m[0]*m[7]*m[9]   + m[4]*m[1]*m[11] - m[4]*m[3]*m[9]  - m[8]*m[1]*m[7]   + m[8]*m[3]*m[5];
    inv[15] =  m[0]*m[5]*m[10]  - m[0]*m[6]*m[9]   - m[4]*m[1]*m[10] + m[4]*m[2]*m[9]  + m[8]*m[1]*m[6]   - m[8]*m[2]*m[5];
    float det = m[0]*inv[0] + m[1]*inv[4] + m[2]*inv[8] + m[3]*inv[12];
    det = 1.0f / det;
#pragma unroll
    for (int i = 0; i < 16; ++i) o[i] = inv[i] * det;
}

// ---------------------------------------------------------------------------
// Kernel 0: per (b, pair) relative transform M = E_t * inv(E_s); zero accum.
// ---------------------------------------------------------------------------
__global__ void setup_kernel(const float* __restrict__ E,  // (B, N, 4, 4)
                             float* __restrict__ M,        // (B, 12, 16)
                             float* __restrict__ acc)      // (24)
{
    const int t = threadIdx.x;
    if (t < 24) {
        acc[t] = 0.0f;
        const int b = t / NPAIRS;
        const int p = t % NPAIRS;
        const int s  = c_ps[p];
        const int tv = c_pt[p];
        const float* Es = E + (size_t)(b * NN + s)  * 16;
        const float* Et = E + (size_t)(b * NN + tv) * 16;
        float inv[16];
        inv4x4(Es, inv);
        float* Mo = M + (size_t)(b * NPAIRS + p) * 16;
#pragma unroll
        for (int i = 0; i < 4; ++i)
#pragma unroll
            for (int j = 0; j < 4; ++j) {
                float acc_ij = 0.0f;
#pragma unroll
                for (int k = 0; k < 4; ++k)
                    acc_ij += Et[i * 4 + k] * inv[k * 4 + j];
                Mo[i * 4 + j] = acc_ij;
            }
    }
}

// ---------------------------------------------------------------------------
// Kernel 2: per-pair reprojection Huber loss
// wave32 shfl butterfly -> 8 partials in LDS -> one atomic per block
// grid: (B*H*W/256, 12)
// ---------------------------------------------------------------------------
__global__ __launch_bounds__(256) void pairwise_loss(
    const float* __restrict__ depth,   // (B, N, 1, H, W)
    const float* __restrict__ intr,    // (B, N, 3, 3)
    const float* __restrict__ M,       // (B, 12, 16)
    float* __restrict__ acc)           // [0..11]=hsum, [12..23]=msum
{
    const int p  = blockIdx.y;
    const int s  = c_ps[p];
    const int tv = c_pt[p];

    const int tid  = threadIdx.x;
    const int lane = tid & 31;
    const int wave = tid >> 5;

    const int idx = blockIdx.x * 256 + tid;           // 0 .. B*H*W-1
    const int b   = idx / (HH * WW);
    const int rem = idx % (HH * WW);
    const int y   = rem / WW;
    const int x   = rem % WW;

    const float* Ks = intr + (size_t)(b * NN + s)  * 9;
    const float* Kt = intr + (size_t)(b * NN + tv) * 9;
    const float fxs = Ks[0], cxs = Ks[2], fys = Ks[4], cys = Ks[5];
    const float fxt = Kt[0], cxt = Kt[2], fyt = Kt[4], cyt = Kt[5];

    const float Z = depth[((size_t)(b * NN + s) * HH + y) * WW + x];
    const float X = ((float)x - cxs) * Z / fxs;
    const float Y = ((float)y - cys) * Z / fys;

    const float* Mm = M + (size_t)(b * NPAIRS + p) * 16;
    const float Xc = Mm[0]*X + Mm[1]*Y + Mm[2]*Z  + Mm[3];
    const float Yc = Mm[4]*X + Mm[5]*Y + Mm[6]*Z  + Mm[7];
    const float Zt = Mm[8]*X + Mm[9]*Y + Mm[10]*Z + Mm[11];

    const float xp = Xc * fxt / (Zt + 1e-6f) + cxt;
    const float yp = Yc * fyt / (Zt + 1e-6f) + cyt;
    const float mask = (Zt > 0.1f && Zt < 1000.0f &&
                        xp >= 0.0f && xp < (float)WW &&
                        yp >= 0.0f && yp < (float)HH) ? 1.0f : 0.0f;

    // bilinear sample of target depth (zero outside)
    const float* dt = depth + (size_t)(b * NN + tv) * HH * WW;
    const float x0f = floorf(xp), y0f = floorf(yp);
    const int x0 = (int)x0f, y0 = (int)y0f;
    const float wx = xp - x0f, wy = yp - y0f;

    auto gather = [&](int xi, int yi) -> float {
        if (xi < 0 || xi >= WW || yi < 0 || yi >= HH) return 0.0f;
        return dt[yi * WW + xi];
    };
    const float v00 = gather(x0,     y0);
    const float v10 = gather(x0 + 1, y0);
    const float v01 = gather(x0,     y0 + 1);
    const float v11 = gather(x0 + 1, y0 + 1);
    const float samp = v00 * (1.0f - wx) * (1.0f - wy) + v10 * wx * (1.0f - wy)
                     + v01 * (1.0f - wx) * wy          + v11 * wx * wy;

    const float diff = fabsf(Zt - samp);
    const float h = diff < DELTA_H ? 0.5f * diff * diff
                                   : DELTA_H * (diff - 0.5f * DELTA_H);

    // ---- wave32 butterfly reduction (no LDS traffic) ----
    float vm = mask;
    float vh = h * mask;
#pragma unroll
    for (int o = 16; o >= 1; o >>= 1) {
        vm += __shfl_xor(vm, o, 32);
        vh += __shfl_xor(vh, o, 32);
    }

    __shared__ float pm[8];
    __shared__ float ph[8];
    if (lane == 0) { pm[wave] = vm; ph[wave] = vh; }
    __syncthreads();

    if (tid == 0) {
        float tm = 0.0f, th = 0.0f;
#pragma unroll
        for (int i = 0; i < 8; ++i) { tm += pm[i]; th += ph[i]; }
        atomicAdd(&acc[p],      th);   // hsum
        atomicAdd(&acc[12 + p], tm);   // msum
    }
}

// ---------------------------------------------------------------------------
// Kernel 3: gate pairs, average, scale by sigmoid(cw), write scalar
// ---------------------------------------------------------------------------
__global__ void finalize_kernel(const float* __restrict__ acc,
                                const float* __restrict__ cw,
                                float* __restrict__ out)
{
    if (threadIdx.x == 0 && blockIdx.x == 0) {
        const float min_count = 0.3f * (float)BB * (float)HH * (float)WW;
        float total = 0.0f, count = 0.0f;
#pragma unroll
        for (int p = 0; p < NPAIRS; ++p) {
            const float ms = acc[12 + p];
            if (ms > min_count) {
                total += acc[p] / (ms + 1e-6f);
                count += 1.0f;
            }
        }
        total = (count > 0.0f) ? total / count : 0.0f;
        const float sw = 1.0f / (1.0f + __expf(-cw[0]));
        out[(size_t)BB * NN * HH * WW] = total * sw;
    }
}

// ---------------------------------------------------------------------------
extern "C" void kernel_launch(void* const* d_in, const int* in_sizes, int n_in,
                              void* d_out, int out_size, void* d_ws, size_t ws_size,
                              hipStream_t stream)
{
    const float* depth = (const float*)d_in[0];   // (B,N,1,H,W)
    const float* E     = (const float*)d_in[1];   // (B,N,4,4)
    const float* intr  = (const float*)d_in[2];   // (B,N,3,3)
    const float* w1    = (const float*)d_in[3];   // (32,1,3,3)
    const float* b1    = (const float*)d_in[4];   // (32)
    const float* w2    = (const float*)d_in[5];   // (1,32,1,1)
    const float* b2    = (const float*)d_in[6];   // (1)
    const float* cw    = (const float*)d_in[7];   // scalar

    float* conf = (float*)d_out;                  // (B,N,1,H,W) + 1 scalar

    float* Mws = (float*)d_ws;                    // 24 * 16 floats
    float* acc = Mws + BB * NPAIRS * 16;          // 24 floats

    setup_kernel<<<1, 32, 0, stream>>>(E, Mws, acc);

    const int convBlocks = (BB * NN * HH * WW) / 128;  // 8 waves * 16 px
    conf_conv_wmma<<<convBlocks, 256, 0, stream>>>(depth, w1, b1, w2, b2, conf);

    dim3 grid((BB * HH * WW) / 256, NPAIRS);
    pairwise_loss<<<grid, 256, 0, stream>>>(depth, intr, Mws, acc);

    finalize_kernel<<<1, 32, 0, stream>>>(acc, cw, conf);
}
